// BasicBlock_1w1a_76424648065832
// MI455X (gfx1250) — 16-bit and 8-bit matrix operand order repaired and checked by reading, compiled, not run on hardware
//
#include <hip/hip_runtime.h>
#include <hip/hip_bf16.h>

typedef __attribute__((ext_vector_type(8)))  int      v8i;
typedef __attribute__((ext_vector_type(8)))  float    v8f;
typedef __attribute__((ext_vector_type(16))) _Float16 v16h;


__device__ __forceinline__ void async_copy16(void* lds_dst, const void* gsrc) {
  unsigned la = (unsigned)(unsigned long long)lds_dst;
  unsigned long long ga = (unsigned long long)gsrc;
  asm volatile("global_load_async_to_lds_b128 %0, %1, off"
               :: "v"(la), "v"(ga) : "memory");
}
__device__ __forceinline__ void async_wait0() {
  asm volatile("s_wait_asynccnt 0x0" ::: "memory");
}

__global__ void k_binw(const float* __restrict__ w, signed char* __restrict__ wsgn,
                       float* __restrict__ alpha) {
  int o = blockIdx.x;
  int c = threadIdx.x;
  __shared__ float red[256];
  float s = 0.f;
  const float* wrow = w + (size_t)(o * 256 + c) * 9;
  for (int t = 0; t < 9; ++t) {
    float v = wrow[t];
    s += fabsf(v);
    signed char sg = (v > 0.f) ? 1 : ((v < 0.f) ? -1 : 0);
    wsgn[(size_t)(t * 256 + o) * 256 + c] = sg;
  }
  red[c] = s;
  __syncthreads();
  for (int st = 128; st > 0; st >>= 1) {
    if (c < st) red[c] += red[c + st];
    __syncthreads();
  }
  if (c == 0) alpha[o] = red[0] * (1.f / 2304.f);
}

__global__ void k_binpool(const float* __restrict__ x, signed char* __restrict__ As,
                          float* __restrict__ p) {
  int nc = blockIdx.x;
  int n = nc >> 8, c = nc & 255;
  int t = threadIdx.x;
  __shared__ float blk[16];
  if (t < 16) blk[t] = 0.f;
  __syncthreads();
  const float* plane = x + (size_t)nc * 1024;
  float part = 0.f;
  int px0 = t * 4;
  int h = px0 >> 5;
  #pragma unroll
  for (int i = 0; i < 4; ++i) {
    int px = px0 + i;
    int wc = px & 31;
    float v = plane[px];
    part += v;
    signed char sg = (v > 0.f) ? 1 : ((v < 0.f) ? -1 : 0);
    As[(size_t)((n * 32 + h) * 32 + wc) * 256 + c] = sg;
  }
  int bi = h >> 3, bj = (px0 & 31) >> 3;
  atomicAdd(&blk[bi * 4 + bj], part);
  __syncthreads();
  if (t < 16) p[(size_t)nc * 16 + t] = blk[t] * (1.f / 64.f);
}

__global__ void k_dada_gemm(const float* __restrict__ p, const float* __restrict__ dw,
                            float* __restrict__ y) {
  int bid = blockIdx.x;
  int mt = bid >> 4, nt = bid & 15;
  int lane = threadIdx.x;
  int ml = lane & 15, hi = lane >> 4;
  int m = mt * 16 + ml;
  int nidx = m >> 4, ij = m & 15;
  int o = nt * 16 + ml;
  v8f acc = {};
  for (int k0 = 0; k0 < 256; k0 += 32) {
    v16h A, B;
    #pragma unroll
    for (int e = 0; e < 16; ++e) {
      int Ka = ((e >> 3) * 16) + hi * 8 + (e & 7);
      A[e] = (_Float16)p[((size_t)nidx * 256 + (k0 + Ka)) * 16 + ij];
      int Kb = ((e >> 3) * 16) + hi * 8 + (e & 7);
      B[e] = (_Float16)dw[(size_t)o * 256 + k0 + Kb];
    }
    acc = __builtin_amdgcn_wmma_f32_16x16x32_f16(false, A, false, B,
                                                 (short)0, acc, false, false);
  }
  #pragma unroll
  for (int rr = 0; rr < 8; ++rr) {
    int mrow = mt * 16 + rr + hi * 8;
    y[(size_t)mrow * 256 + o] = acc[rr];
  }
}

__global__ void k_dada_stats(const float* __restrict__ y, const float* __restrict__ g,
                             const float* __restrict__ b, float* __restrict__ sig) {
  int o = blockIdx.x;
  int t = threadIdx.x;
  __shared__ float rs_[256], rq_[256];
  float s = 0.f, q = 0.f;
  for (int i = 0; i < 4; ++i) {
    float v = y[(size_t)(t * 4 + i) * 256 + o];
    s += v; q += v * v;
  }
  rs_[t] = s; rq_[t] = q;
  __syncthreads();
  for (int st = 128; st > 0; st >>= 1) {
    if (t < st) { rs_[t] += rs_[t + st]; rq_[t] += rq_[t + st]; }
    __syncthreads();
  }
  __shared__ float mu, rstd;
  if (t == 0) {
    float m = rs_[0] * (1.f / 1024.f);
    float var = rq_[0] * (1.f / 1024.f) - m * m;
    mu = m; rstd = rsqrtf(var + 1e-5f);
  }
  __syncthreads();
  if (t < 64) {
    float yb = 0.f;
    for (int ij = 0; ij < 16; ++ij) yb += y[(size_t)(t * 16 + ij) * 256 + o];
    yb *= (1.f / 16.f);
    float mv = (yb - mu) * rstd * g[o] + b[o];
    sig[t * 256 + o] = 1.f / (1.f + expf(-mv));
  }
}

__global__ __launch_bounds__(256) void k_conv(
    const signed char* __restrict__ As, const signed char* __restrict__ wsgn,
    const float* __restrict__ alpha, const float* __restrict__ sig,
    const float* __restrict__ aprelu, float* __restrict__ T) {
  extern __shared__ char smem[];
  signed char* sA = (signed char*)smem;
  signed char* sB = (signed char*)smem + 34816;
  int bid = blockIdx.x;
  int n = bid >> 4, h0 = (bid & 15) * 2;
  int tid = threadIdx.x;
  const int4 z4 = make_int4(0, 0, 0, 0);

  if (tid < 128) {
    int r = tid >> 5, cpad = (tid >> 4) & 1, q = tid & 15;
    int colB = cpad ? 33 : 0;
    ((int4*)(sA + (r * 34 + colB) * 256))[q] = z4;
  }
  for (int r = 0; r < 4; ++r) {
    int hin = h0 - 1 + r;
    if (hin >= 0 && hin < 32) continue;
    int4* dst = (int4*)(sA + (r * 34 + 1) * 256);
    for (int i = tid; i < 512; i += 256) dst[i] = z4;
  }
  for (int r = 0; r < 4; ++r) {
    int hin = h0 - 1 + r;
    if (hin < 0 || hin >= 32) continue;
    const signed char* src = As + (size_t)((n * 32 + hin) * 32) * 256;
    signed char* dst = sA + (r * 34 + 1) * 256;
    for (int i = tid; i < 512; i += 256)
      async_copy16(dst + i * 16, src + i * 16);
  }

  int lane = tid & 31, wv = tid >> 5;
  int ml = lane & 15, hi = lane >> 4;
  int mt = wv & 3, nh = wv >> 2;
  int rrow = mt >> 1, col = (mt & 1) * 16 + ml;

  v8i acc[8] = {};

  for (int c0 = 0; c0 < 256; c0 += 64) {
    __syncthreads();
    for (int i = tid; i < 9216; i += 256) {
      int k4 = i & 3, o = (i >> 2) & 255, tp = i >> 10;
      const signed char* src = wsgn + (size_t)(tp * 256 + o) * 256 + c0 + k4 * 16;
      async_copy16(sB + (size_t)i * 16, src);
    }
    if (c0 < 192) __builtin_prefetch(wsgn + c0 + 64, 0, 0);
    async_wait0();
    __syncthreads();

    for (int tap = 0; tap < 9; ++tap) {
      int kh = tap / 3, kw = tap % 3;
      int abase = ((rrow + kh) * 34 + (col + kw)) * 256 + c0 + hi * 16;
      v8i a;
      #pragma unroll
      for (int vp = 0; vp < 4; ++vp) {
        int2 tt = *(const int2*)(sA + abase + (vp >> 1) * 32 + (vp & 1) * 8);
        a[2 * vp] = tt.x; a[2 * vp + 1] = tt.y;
      }

      #pragma unroll
      for (int j = 0; j < 8; ++j) {
        int nt = nh * 8 + j;
        const signed char* bb = sB + tap * 16384 + (nt * 16 + ml) * 64 + hi * 16;
        int4 t0 = *(const int4*)(bb);
        int4 t1 = *(const int4*)(bb + 32);
        v8i b = { t0.x, t0.y, t0.z, t0.w, t1.x, t1.y, t1.z, t1.w };
        acc[j] = __builtin_amdgcn_wmma_i32_16x16x64_iu8(true, a, true, b,
                                                        acc[j], false, false);
      }
    }
  }

  #pragma unroll
  for (int j = 0; j < 8; ++j) {
    int o = (nh * 8 + j) * 16 + ml;
    float sc = alpha[o] * sig[n * 256 + o];
    float ap = aprelu[o];
    #pragma unroll
    for (int rr = 0; rr < 8; ++rr) {
      int wcol = (mt & 1) * 16 + rr + hi * 8;
      int h = h0 + rrow;
      float v = sc * (float)acc[j][rr];
      v = (v >= 0.f) ? v : ap * v;
      T[(size_t)((n * 256 + o) * 32 + h) * 32 + wcol] = v;
    }
  }
}

__global__ void k_chstats(const float* __restrict__ T, float* __restrict__ st) {
  int bid = blockIdx.x;
  int o = bid & 255;
  int t = threadIdx.x;
  __shared__ float rs_[256], rq_[256];
  const float* base = T + (size_t)bid * 1024;
  float s = 0.f, q = 0.f;
  for (int i = 0; i < 4; ++i) { float v = base[t * 4 + i]; s += v; q += v * v; }
  rs_[t] = s; rq_[t] = q;
  __syncthreads();
  for (int stp = 128; stp > 0; stp >>= 1) {
    if (t < stp) { rs_[t] += rs_[t + stp]; rq_[t] += rq_[t + stp]; }
    __syncthreads();
  }
  if (t == 0) { atomicAdd(&st[o], rs_[0]); atomicAdd(&st[256 + o], rq_[0]); }
}

__global__ void k_finalize(const float* __restrict__ T, const float* __restrict__ st,
                           const float* __restrict__ g, const float* __restrict__ b,
                           const float* __restrict__ resid, float* __restrict__ dst) {
  size_t idx = (size_t)blockIdx.x * 256 + threadIdx.x;
  int o = (int)((idx >> 10) & 255);
  const float inv = 1.f / 65536.f;
  float mu = st[o] * inv;
  float var = st[256 + o] * inv - mu * mu;
  float v = (T[idx] - mu) * rsqrtf(var + 1e-5f) * g[o] + b[o] + resid[idx];
  dst[idx] = v;
}

extern "C" void kernel_launch(void* const* d_in, const int* in_sizes, int n_in,
                              void* d_out, int out_size, void* d_ws, size_t ws_size,
                              hipStream_t stream) {
  (void)in_sizes; (void)n_in; (void)out_size; (void)ws_size;
  const float* x   = (const float*)d_in[0];
  const float* w1  = (const float*)d_in[1];
  const float* g1  = (const float*)d_in[2];
  const float* b1  = (const float*)d_in[3];
  const float* a1  = (const float*)d_in[4];
  const float* dw1 = (const float*)d_in[5];
  const float* dg1 = (const float*)d_in[6];
  const float* db1 = (const float*)d_in[7];
  const float* w2  = (const float*)d_in[8];
  const float* g2  = (const float*)d_in[9];
  const float* b2  = (const float*)d_in[10];
  const float* a2  = (const float*)d_in[11];
  const float* dw2 = (const float*)d_in[12];
  const float* dg2 = (const float*)d_in[13];
  const float* db2 = (const float*)d_in[14];

  char* ws = (char*)d_ws;
  signed char* As    = (signed char*)(ws);
  float*       X1    = (float*)(ws + 16777216);
  float*       p     = (float*)(ws + 83886080);
  float*       y     = (float*)(ws + 84934656);
  float*       sig   = (float*)(ws + 85983232);
  float*       alpha = (float*)(ws + 86048768);
  signed char* wsgn  = (signed char*)(ws + 86049792);
  float*       st    = (float*)(ws + 86639616);
  float*       T     = (float*)d_out;

  const size_t shmem = 34816 + 147456;

  k_binw      <<<256,   256, 0,     stream>>>(w1, wsgn, alpha);
  k_binpool   <<<16384, 256, 0,     stream>>>(x, As, p);
  k_dada_gemm <<<1024,  32,  0,     stream>>>(p, dw1, y);
  k_dada_stats<<<256,   256, 0,     stream>>>(y, dg1, db1, sig);
  k_conv      <<<1024,  256, shmem, stream>>>(As, wsgn, alpha, sig, a1, T);
  hipMemsetAsync(st, 0, 2048, stream);
  k_chstats   <<<16384, 256, 0,     stream>>>(T, st);
  k_finalize  <<<65536, 256, 0,     stream>>>(T, st, g1, b1, x, X1);

  k_binw      <<<256,   256, 0,     stream>>>(w2, wsgn, alpha);
  k_binpool   <<<16384, 256, 0,     stream>>>(X1, As, p);
  k_dada_gemm <<<1024,  32,  0,     stream>>>(p, dw2, y);
  k_dada_stats<<<256,   256, 0,     stream>>>(y, dg2, db2, sig);
  k_conv      <<<1024,  256, shmem, stream>>>(As, wsgn, alpha, sig, a2, T);
  hipMemsetAsync(st, 0, 2048, stream);
  k_chstats   <<<16384, 256, 0,     stream>>>(T, st);
  k_finalize  <<<65536, 256, 0,     stream>>>(T, st, g2, b2, X1, (float*)d_out);
}
